// KGramMLPSeqModel_91018946937243
// MI455X (gfx1250) — matrix operand order repaired and checked by reading, not compiled, not run on hardware
//
#include <hip/hip_runtime.h>
#include <hip/hip_bf16.h>

// ---------------------------------------------------------------------------
// KGram MLP seq model for MI455X (gfx1250, wave32, WMMA 16x16x32 bf16)
//   stage0: transpose-convert w1 (768x256 f32) -> w1t (256x768 bf16)
//           transpose-convert w2 (256x32000 f32) -> w2t (32000x256 bf16)
//   stage1: h = silu(gathered_embed @ w1 + b1)  -> bf16 workspace (8192x256)
//   stage2: logits = h @ w2 + b2                -> f32 d_out (8192x32000)
// ---------------------------------------------------------------------------

typedef __bf16 v16bf __attribute__((ext_vector_type(16)));
typedef float  v8f   __attribute__((ext_vector_type(8)));

#define SEQ     2048
#define BATCH   4
#define M_TOT   (SEQ * BATCH)   // 8192
#define E_DIM   256
#define KGRAM   3
#define KE      768             // KGRAM * E_DIM
#define H_DIM   256
#define VOCAB   32000

// Workspace layout (bytes):
//   [0,                16384000)  w2t  : 32000*256 bf16
//   [16384000,         16777216)  w1t  : 256*768  bf16
//   [16777216,         20971520)  hbf  : 8192*256 bf16
#define WS_W2T_OFF  0
#define WS_W1T_OFF  16384000
#define WS_H_OFF    16777216

// ---- tiled transpose + f32->bf16: in[R][C] (f32) -> out[C][R] (bf16) ------
__global__ __launch_bounds__(256) void transpose_cvt_kernel(
    const float* __restrict__ in, __bf16* __restrict__ out, int R, int C) {
  __shared__ float tile[32][33];
  const int c0 = blockIdx.x * 32;
  const int r0 = blockIdx.y * 32;
  const int tx = threadIdx.x & 31;
  const int ty = threadIdx.x >> 5;  // 0..7
#pragma unroll
  for (int i = 0; i < 32; i += 8)
    tile[ty + i][tx] = in[(size_t)(r0 + ty + i) * C + (c0 + tx)];
  __syncthreads();
#pragma unroll
  for (int i = 0; i < 32; i += 8)
    out[(size_t)(c0 + ty + i) * R + (r0 + tx)] = (__bf16)tile[tx][ty + i];
}

// ---- fragment loaders -----------------------------------------------------
// bf16 fragment: two 16B contiguous chunks (ISA VGPR layouts, CDNA5 ISA).
__device__ __forceinline__ v16bf load_frag_bf16(const __bf16* p0,
                                                const __bf16* p1) {
  union { uint4 u[2]; v16bf v; } f;
  f.u[0] = *(const uint4*)p0;
  f.u[1] = *(const uint4*)p1;
  return f.v;
}

// A-fragment from f32 source rows (convert to bf16 in registers).
__device__ __forceinline__ v16bf load_a_cvt_f32(const float* p0,
                                                const float* p1) {
  float4 x0 = *(const float4*)(p0);
  float4 x1 = *(const float4*)(p0 + 4);
  float4 y0 = *(const float4*)(p1);
  float4 y1 = *(const float4*)(p1 + 4);
  v16bf a;
  a[0] = (__bf16)x0.x;  a[1] = (__bf16)x0.y;
  a[2] = (__bf16)x0.z;  a[3] = (__bf16)x0.w;
  a[4] = (__bf16)x1.x;  a[5] = (__bf16)x1.y;
  a[6] = (__bf16)x1.z;  a[7] = (__bf16)x1.w;
  a[8]  = (__bf16)y0.x; a[9]  = (__bf16)y0.y;
  a[10] = (__bf16)y0.z; a[11] = (__bf16)y0.w;
  a[12] = (__bf16)y1.x; a[13] = (__bf16)y1.y;
  a[14] = (__bf16)y1.z; a[15] = (__bf16)y1.w;
  return a;
}

// ---- stage 1: h = silu(x @ w1 + b1), x gathered from embed ---------------
// grid = (H_DIM/128, M_TOT/128), block = 256 (8 waves: 4 in M x 2 in N).
__global__ __launch_bounds__(256) void kgram_stage1(
    const int* __restrict__ tokens,        // (S, B) int32
    const float* __restrict__ embed,       // (V, 256) f32
    const __bf16* __restrict__ w1t,        // (256 n, 768 k) bf16
    const float* __restrict__ b1,          // (256,)
    __bf16* __restrict__ hout) {           // (8192, 256) bf16
  const int lane  = threadIdx.x & 31;
  const int wave  = threadIdx.x >> 5;
  const int waveM = wave & 3;
  const int waveN = wave >> 2;
  const int mBase = blockIdx.y * 128 + waveM * 32;
  const int nBase = blockIdx.x * 128 + waveN * 64;
  const int l15   = lane & 15;
  const int hi    = lane >> 4;     // 0 | 1
  const int ksel  = hi * 8;        // A-frag K sub-select
  const int khalf = hi * 16;       // unused; B-frag also uses ksel

  v8f acc[2][4];
  const v8f vzero = {0.f, 0.f, 0.f, 0.f, 0.f, 0.f, 0.f, 0.f};
#pragma unroll
  for (int mt = 0; mt < 2; ++mt)
#pragma unroll
    for (int nt = 0; nt < 4; ++nt) acc[mt][nt] = vzero;

  int rows[2], ss[2], bb[2];
#pragma unroll
  for (int mt = 0; mt < 2; ++mt) {
    rows[mt] = mBase + mt * 16 + l15;
    ss[mt]   = rows[mt] >> 2;     // s = m / BATCH
    bb[mt]   = rows[mt] & 3;      // b = m % BATCH
  }

  const __bf16* brow[4];
#pragma unroll
  for (int nt = 0; nt < 4; ++nt)
    brow[nt] = w1t + (size_t)(nBase + nt * 16 + l15) * KE;

  for (int kt = 0; kt < KGRAM; ++kt) {
    const float* arow[2];
#pragma unroll
    for (int mt = 0; mt < 2; ++mt) {
      int tix = ss[mt] - KGRAM + kt;                 // padded window
      int tok = (tix >= 0) ? tokens[tix * BATCH + bb[mt]] : 0;
      arow[mt] = embed + (size_t)tok * E_DIM;
    }
#pragma unroll
    for (int kc = 0; kc < 8; ++kc) {
      const int e0 = kc * 32 + ksel;
      v16bf afr[2], bfr[4];
#pragma unroll
      for (int mt = 0; mt < 2; ++mt)
        afr[mt] = load_a_cvt_f32(arow[mt] + e0, arow[mt] + e0 + 16);
      const int kg = kt * E_DIM + kc * 32 + ksel;
#pragma unroll
      for (int nt = 0; nt < 4; ++nt)
        bfr[nt] = load_frag_bf16(brow[nt] + kg, brow[nt] + kg + 16);
#pragma unroll
      for (int mt = 0; mt < 2; ++mt)
#pragma unroll
        for (int nt = 0; nt < 4; ++nt)
          acc[mt][nt] = __builtin_amdgcn_wmma_f32_16x16x32_bf16(
              false, afr[mt], false, bfr[nt], (short)0, acc[mt][nt],
              false, false);
    }
  }

  // epilogue: bias + silu, store bf16
#pragma unroll
  for (int nt = 0; nt < 4; ++nt) {
    const int col = nBase + nt * 16 + l15;
    const float bias = b1[col];
#pragma unroll
    for (int mt = 0; mt < 2; ++mt) {
#pragma unroll
      for (int r = 0; r < 8; ++r) {
        const int row = mBase + mt * 16 + hi * 8 + r;
        float x = acc[mt][nt][r] + bias;
        float h = x / (1.0f + __expf(-x));
        hout[(size_t)row * H_DIM + col] = (__bf16)h;
      }
    }
  }
}

// ---- stage 2: logits = h @ w2 + b2 ---------------------------------------
// grid = (VOCAB/128, M_TOT/128), block = 256 (8 waves: 4 in M x 2 in N).
__global__ __launch_bounds__(256) void kgram_stage2(
    const __bf16* __restrict__ hbf,        // (8192, 256) bf16
    const __bf16* __restrict__ w2t,        // (32000 n, 256 k) bf16
    const float* __restrict__ b2,          // (32000,)
    float* __restrict__ out) {             // (8192, 32000) f32
  const int lane  = threadIdx.x & 31;
  const int wave  = threadIdx.x >> 5;
  const int waveM = wave & 3;
  const int waveN = wave >> 2;
  const int mBase = blockIdx.y * 128 + waveM * 32;
  const int nBase = blockIdx.x * 128 + waveN * 64;
  const int l15   = lane & 15;
  const int hi    = lane >> 4;
  const int ksel  = hi * 8;
  const int khalf = hi * 16;

  v8f acc[2][4];
  const v8f vzero = {0.f, 0.f, 0.f, 0.f, 0.f, 0.f, 0.f, 0.f};
#pragma unroll
  for (int mt = 0; mt < 2; ++mt)
#pragma unroll
    for (int nt = 0; nt < 4; ++nt) acc[mt][nt] = vzero;

  const __bf16* arow[2];
#pragma unroll
  for (int mt = 0; mt < 2; ++mt)
    arow[mt] = hbf + (size_t)(mBase + mt * 16 + l15) * H_DIM;
  const __bf16* brow[4];
#pragma unroll
  for (int nt = 0; nt < 4; ++nt)
    brow[nt] = w2t + (size_t)(nBase + nt * 16 + l15) * H_DIM;

#pragma unroll
  for (int kk = 0; kk < H_DIM; kk += 32) {
    v16bf afr[2], bfr[4];
#pragma unroll
    for (int mt = 0; mt < 2; ++mt)
      afr[mt] = load_frag_bf16(arow[mt] + kk + ksel, arow[mt] + kk + 16 + ksel);
#pragma unroll
    for (int nt = 0; nt < 4; ++nt)
      bfr[nt] = load_frag_bf16(brow[nt] + kk + ksel, brow[nt] + kk + 16 + ksel);
#pragma unroll
    for (int mt = 0; mt < 2; ++mt)
#pragma unroll
      for (int nt = 0; nt < 4; ++nt)
        acc[mt][nt] = __builtin_amdgcn_wmma_f32_16x16x32_bf16(
            false, afr[mt], false, bfr[nt], (short)0, acc[mt][nt],
            false, false);
  }

  // epilogue: bias + f32 store (streaming output is the roofline bound)
#pragma unroll
  for (int nt = 0; nt < 4; ++nt) {
    const int col = nBase + nt * 16 + l15;
    const float bias = b2[col];
#pragma unroll
    for (int mt = 0; mt < 2; ++mt) {
#pragma unroll
      for (int r = 0; r < 8; ++r) {
        const int row = mBase + mt * 16 + hi * 8 + r;
        out[(size_t)row * VOCAB + col] = acc[mt][nt][r] + bias;
      }
    }
  }
}

// ---------------------------------------------------------------------------
extern "C" void kernel_launch(void* const* d_in, const int* in_sizes, int n_in,
                              void* d_out, int out_size, void* d_ws,
                              size_t ws_size, hipStream_t stream) {
  const int*   tokens = (const int*)d_in[0];    // (S, B)
  const float* embed  = (const float*)d_in[1];  // (V, E)
  const float* w1     = (const float*)d_in[2];  // (K*E, H)
  const float* b1     = (const float*)d_in[3];  // (H,)
  const float* w2     = (const float*)d_in[4];  // (H, V)
  const float* b2     = (const float*)d_in[5];  // (V,)
  float* out          = (float*)d_out;          // (S, B, V)

  char* ws = (char*)d_ws;
  __bf16* w2t = (__bf16*)(ws + WS_W2T_OFF);     // (VOCAB, H_DIM)
  __bf16* w1t = (__bf16*)(ws + WS_W1T_OFF);     // (H_DIM, KE)
  __bf16* hbf = (__bf16*)(ws + WS_H_OFF);       // (M_TOT, H_DIM)

  // stage 0: weight transpose + bf16 convert (L2-resident, done per call)
  transpose_cvt_kernel<<<dim3(VOCAB / 32, H_DIM / 32), 256, 0, stream>>>(
      w2, w2t, H_DIM, VOCAB);
  transpose_cvt_kernel<<<dim3(H_DIM / 32, KE / 32), 256, 0, stream>>>(
      w1, w1t, KE, H_DIM);

  // stage 1: gather + GEMM1 + silu
  kgram_stage1<<<dim3(H_DIM / 128, M_TOT / 128), 256, 0, stream>>>(
      tokens, embed, w1t, b1, hbf);

  // stage 2: GEMM2 + bias (store-bound; ~45 us floor at 23.3 TB/s)
  kgram_stage2<<<dim3(VOCAB / 128, M_TOT / 128), 256, 0, stream>>>(
      hbf, w2t, b2, out);
}
